// MASSVariationalDist_34333968564439
// MI455X (gfx1250) — matrix operand order repaired and checked by reading, not compiled, not run on hardware
//
#include <hip/hip_runtime.h>
#include <hip/hip_bf16.h>
#include <math.h>

typedef __bf16 bf16_t;
typedef __bf16 v16bf __attribute__((ext_vector_type(16)));
typedef __bf16 v8bf  __attribute__((ext_vector_type(8)));
typedef float  v8f   __attribute__((ext_vector_type(8)));
typedef int    v4i   __attribute__((ext_vector_type(4)));

#define B_DIM   2048
#define C_DIM   100
#define K_DIM   8
#define CK_DIM  800          // C*K
#define D_DIM   64
#define BTILE   128
#define LOG2PI  1.8378770664093453f

// -------- gfx1250 async global->LDS path (guarded; falls back to uint4 copy) --
#if defined(__has_builtin)
#if __has_builtin(__builtin_amdgcn_global_load_async_to_lds_b128) && \
    __has_builtin(__builtin_amdgcn_s_wait_asynccnt)
#define USE_ASYNC_LDS 1
#endif
#endif
#ifndef USE_ASYNC_LDS
#define USE_ASYNC_LDS 0
#endif

#if USE_ASYNC_LDS
typedef __attribute__((address_space(1))) v4i as1_v4i;   // global
typedef __attribute__((address_space(3))) v4i as3_v4i;   // LDS

__device__ inline void async_copy16(const void* gsrc, void* ldst) {
    __builtin_amdgcn_global_load_async_to_lds_b128(
        (as1_v4i*)(void*)gsrc,
        (as3_v4i*)ldst,
        0, 0);
}
#endif

// ---------------------------------------------------------------------------
// Kernel 0: split representation X [B,64] f32 into bf16 hi/lo pair
// ---------------------------------------------------------------------------
__global__ void splitx_kernel(const float* __restrict__ x,
                              bf16_t* __restrict__ xh, bf16_t* __restrict__ xl,
                              int n) {
    int i = blockIdx.x * blockDim.x + threadIdx.x;
    if (i < n) {
        float v = x[i];
        bf16_t h = (bf16_t)v;
        xh[i] = h;
        xl[i] = (bf16_t)(v - (float)h);
    }
}

// ---------------------------------------------------------------------------
// Kernel 1: per (c,k): invert 64x64 lower-triangular L, store Linv as bf16
// hi/lo, compute w = Linv @ mu and logdet = sum log|diag L|.
// One block of 64 threads per (c,k); thread j owns column j of Linv.
// ---------------------------------------------------------------------------
__global__ void __launch_bounds__(64)
prep_kernel(const float* __restrict__ scale_tril, const float* __restrict__ loc,
            bf16_t* __restrict__ LinvH, bf16_t* __restrict__ LinvL,
            float* __restrict__ wv, float* __restrict__ logdet) {
    __shared__ float Ls[D_DIM * D_DIM];
    __shared__ float Zs[D_DIM * D_DIM];
    const int ck = blockIdx.x;
    const int t  = threadIdx.x;   // 0..63
    const float* L = scale_tril + (size_t)ck * D_DIM * D_DIM;

    for (int i = 0; i < D_DIM; ++i) Ls[i * D_DIM + t] = L[i * D_DIM + t];
    __syncthreads();

    // forward substitution: solve L z = e_t  (column t of Linv)
    const int j = t;
    for (int i = 0; i < D_DIM; ++i) {
        if (i < j) { Zs[i * D_DIM + j] = 0.0f; continue; }
        float s = (i == j) ? 1.0f : 0.0f;
        for (int m = j; m < i; ++m) s -= Ls[i * D_DIM + m] * Zs[m * D_DIM + j];
        Zs[i * D_DIM + j] = s / Ls[i * D_DIM + i];
    }
    __syncthreads();

    // bf16 hi/lo split of Linv (row-major [i][j])
    for (int i = 0; i < D_DIM; ++i) {
        float z = Zs[i * D_DIM + t];
        bf16_t h = (bf16_t)z;
        LinvH[(size_t)ck * D_DIM * D_DIM + i * D_DIM + t] = h;
        LinvL[(size_t)ck * D_DIM * D_DIM + i * D_DIM + t] = (bf16_t)(z - (float)h);
    }

    // w[i] = sum_j Linv[i][j] * mu[j]   (Linv is lower triangular)
    const float* mu = loc + (size_t)ck * D_DIM;
    float s = 0.0f;
    for (int jj = 0; jj <= t; ++jj) s += Zs[t * D_DIM + jj] * mu[jj];
    wv[ck * D_DIM + t] = s;

    if (t == 0) {
        float ld = 0.0f;
        for (int i = 0; i < D_DIM; ++i) ld += logf(fabsf(Ls[i * D_DIM + i]));
        logdet[ck] = ld;
    }
}

// ---------------------------------------------------------------------------
// WMMA fragment loaders (contiguous k order, same for A and B, wave32)
// A 16x32 bf16: lane half = lane/16, m = lane&15;
//   e=0..15 -> k = ks*32 + half*16 + e
// B 32x16 bf16: n = lane&15; e=0..15 -> k = ks*32 + half*16 + e
// ---------------------------------------------------------------------------
__device__ inline v16bf load_a_frag(const bf16_t* base, int row, int ks, int half) {
    const bf16_t* p = base + row * D_DIM + ks * 32 + half * 16;
    v8bf lo = *(const v8bf*)p;
    v8bf hi = *(const v8bf*)(p + 8);
    v16bf r;
#pragma unroll
    for (int e = 0; e < 8; ++e) { r[e] = lo[e]; r[e + 8] = hi[e]; }
    return r;
}

__device__ inline v16bf load_b_frag(const bf16_t* base, int n, int ks, int half) {
    const bf16_t* p = base + n * D_DIM + ks * 32 + half * 16;
    return *(const v16bf*)p;
}

// ---------------------------------------------------------------------------
// Kernel 2: for one (c,k) and one 128-row batch tile:
//   U = Xtile @ Linv^T   (bf16 hi/lo x3 WMMA, f32 accumulate)
//   quad[b] = sum_i (U[b,i] - w[i])^2
//   comp_lp[b,ck] = -0.5*quad - logdet - 0.5*D*log2pi
// 256 threads = 8 waves; wave w owns batch rows [w*16, w*16+16).
// LDS tiles filled via gfx1250 async global->LDS (ASYNCcnt) when available.
// ---------------------------------------------------------------------------
__global__ void __launch_bounds__(256)
gemm_kernel(const bf16_t* __restrict__ Xh, const bf16_t* __restrict__ Xl,
            const bf16_t* __restrict__ LinvH, const bf16_t* __restrict__ LinvL,
            const float* __restrict__ wv, const float* __restrict__ logdet,
            float* __restrict__ cl) {
    __shared__ __align__(16) bf16_t sXh[BTILE * D_DIM];
    __shared__ __align__(16) bf16_t sXl[BTILE * D_DIM];
    __shared__ __align__(32) bf16_t sLh[D_DIM * D_DIM];
    __shared__ __align__(32) bf16_t sLl[D_DIM * D_DIM];
    __shared__ float sw[D_DIM];

    const int ck = blockIdx.x;            // 0..799
    const int b0 = blockIdx.y * BTILE;    // batch tile origin
    const int tid = threadIdx.x;

#if USE_ASYNC_LDS
    // async global->LDS: 16B per lane per issue, no VGPR staging
    {
        const char* gX = (const char*)(Xh + (size_t)b0 * D_DIM);     // 16 KB
        const char* gXl = (const char*)(Xl + (size_t)b0 * D_DIM);    // 16 KB
        const char* gLh = (const char*)(LinvH + (size_t)ck * D_DIM * D_DIM); // 8 KB
        const char* gLl = (const char*)(LinvL + (size_t)ck * D_DIM * D_DIM); // 8 KB
#pragma unroll
        for (int i = 0; i < 4; ++i) {   // 256 lanes * 4 * 16B = 16 KB
            int o = (tid + i * 256) * 16;
            async_copy16(gX + o, (char*)sXh + o);
            async_copy16(gXl + o, (char*)sXl + o);
        }
#pragma unroll
        for (int i = 0; i < 2; ++i) {   // 256 lanes * 2 * 16B = 8 KB
            int o = (tid + i * 256) * 16;
            async_copy16(gLh + o, (char*)sLh + o);
            async_copy16(gLl + o, (char*)sLl + o);
        }
    }
    for (int i = tid; i < D_DIM; i += 256) sw[i] = wv[ck * D_DIM + i];
    __builtin_amdgcn_s_wait_asynccnt(0);
#else
    {
        const uint4* g = (const uint4*)(Xh + (size_t)b0 * D_DIM);
        uint4* s = (uint4*)sXh;
        for (int i = tid; i < BTILE * D_DIM / 8; i += 256) s[i] = g[i];
    }
    {
        const uint4* g = (const uint4*)(Xl + (size_t)b0 * D_DIM);
        uint4* s = (uint4*)sXl;
        for (int i = tid; i < BTILE * D_DIM / 8; i += 256) s[i] = g[i];
    }
    {
        const uint4* g = (const uint4*)(LinvH + (size_t)ck * D_DIM * D_DIM);
        uint4* s = (uint4*)sLh;
        for (int i = tid; i < D_DIM * D_DIM / 8; i += 256) s[i] = g[i];
    }
    {
        const uint4* g = (const uint4*)(LinvL + (size_t)ck * D_DIM * D_DIM);
        uint4* s = (uint4*)sLl;
        for (int i = tid; i < D_DIM * D_DIM / 8; i += 256) s[i] = g[i];
    }
    for (int i = tid; i < D_DIM; i += 256) sw[i] = wv[ck * D_DIM + i];
#endif
    __syncthreads();

    const int wave = tid >> 5;
    const int lane = tid & 31;
    const int half = lane >> 4;
    const int mloc = lane & 15;
    const int m0   = wave * 16;

    // preload A fragments for this wave's 16 rows (2 K-steps, hi & lo)
    v16bf ah0 = load_a_frag(sXh, m0 + mloc, 0, half);
    v16bf ah1 = load_a_frag(sXh, m0 + mloc, 1, half);
    v16bf al0 = load_a_frag(sXl, m0 + mloc, 0, half);
    v16bf al1 = load_a_frag(sXl, m0 + mloc, 1, half);

    const float cterm = -logdet[ck] - 0.5f * (float)D_DIM * LOG2PI;

    v8f q;
#pragma unroll
    for (int r = 0; r < 8; ++r) q[r] = 0.0f;

#pragma unroll
    for (int nt = 0; nt < 4; ++nt) {
        v8f acc;
#pragma unroll
        for (int r = 0; r < 8; ++r) acc[r] = 0.0f;

        // K-step 0
        {
            v16bf bh = load_b_frag(sLh, nt * 16 + mloc, 0, half);
            v16bf bl = load_b_frag(sLl, nt * 16 + mloc, 0, half);
            acc = __builtin_amdgcn_wmma_f32_16x16x32_bf16(false, ah0, false, bh,
                                                          (short)0, acc, false, false);
            acc = __builtin_amdgcn_wmma_f32_16x16x32_bf16(false, ah0, false, bl,
                                                          (short)0, acc, false, false);
            acc = __builtin_amdgcn_wmma_f32_16x16x32_bf16(false, al0, false, bh,
                                                          (short)0, acc, false, false);
        }
        // K-step 1
        {
            v16bf bh = load_b_frag(sLh, nt * 16 + mloc, 1, half);
            v16bf bl = load_b_frag(sLl, nt * 16 + mloc, 1, half);
            acc = __builtin_amdgcn_wmma_f32_16x16x32_bf16(false, ah1, false, bh,
                                                          (short)0, acc, false, false);
            acc = __builtin_amdgcn_wmma_f32_16x16x32_bf16(false, ah1, false, bl,
                                                          (short)0, acc, false, false);
            acc = __builtin_amdgcn_wmma_f32_16x16x32_bf16(false, al1, false, bh,
                                                          (short)0, acc, false, false);
        }

        // quad accumulation: D layout -> lane gives N = mloc, VGPR r gives M = r + half*8
        const float wn = sw[nt * 16 + mloc];
#pragma unroll
        for (int r = 0; r < 8; ++r) {
            float z = acc[r] - wn;
            q[r] += z * z;
        }
    }

    // reduce over the N dimension: sum across each 16-lane half
#pragma unroll
    for (int r = 0; r < 8; ++r) {
        float v = q[r];
        v += __shfl_xor(v, 1, 32);
        v += __shfl_xor(v, 2, 32);
        v += __shfl_xor(v, 4, 32);
        v += __shfl_xor(v, 8, 32);
        q[r] = v;
    }

    if (mloc == 0) {
#pragma unroll
        for (int r = 0; r < 8; ++r) {
            int b = b0 + m0 + r + half * 8;
            cl[(size_t)b * CK_DIM + ck] = -0.5f * q[r] + cterm;
        }
    }
}

// ---------------------------------------------------------------------------
// Kernel 3: per batch row b: class_lp[c] = logsumexp_k(comp_lp + log_softmax(mix)),
// then out[b,c] = log_softmax_c(class_lp). One block (128 threads) per b.
// ---------------------------------------------------------------------------
__global__ void __launch_bounds__(128)
finalize_kernel(const float* __restrict__ mix, const float* __restrict__ cl,
                float* __restrict__ out) {
    __shared__ float red[128];
    const int b = blockIdx.x;
    const int c = threadIdx.x;

    float class_lp = -INFINITY;
    if (c < C_DIM) {
        float ml[K_DIM];
        float mmax = -INFINITY;
#pragma unroll
        for (int k = 0; k < K_DIM; ++k) {
            ml[k] = mix[c * K_DIM + k];
            mmax = fmaxf(mmax, ml[k]);
        }
        float msum = 0.0f;
#pragma unroll
        for (int k = 0; k < K_DIM; ++k) msum += expf(ml[k] - mmax);
        const float lsem = mmax + logf(msum);

        float v[K_DIM];
        float vmax = -INFINITY;
#pragma unroll
        for (int k = 0; k < K_DIM; ++k) {
            v[k] = cl[(size_t)b * CK_DIM + c * K_DIM + k] + (ml[k] - lsem);
            vmax = fmaxf(vmax, v[k]);
        }
        float vs = 0.0f;
#pragma unroll
        for (int k = 0; k < K_DIM; ++k) vs += expf(v[k] - vmax);
        class_lp = vmax + logf(vs);
    }

    // block max over classes
    red[c] = class_lp;
    __syncthreads();
    for (int off = 64; off > 0; off >>= 1) {
        if (c < off) red[c] = fmaxf(red[c], red[c + off]);
        __syncthreads();
    }
    const float M = red[0];
    __syncthreads();

    // block sum of exp
    red[c] = (c < C_DIM) ? expf(class_lp - M) : 0.0f;
    __syncthreads();
    for (int off = 64; off > 0; off >>= 1) {
        if (c < off) red[c] += red[c + off];
        __syncthreads();
    }
    const float lse = M + logf(red[0]);

    if (c < C_DIM) out[(size_t)b * C_DIM + c] = class_lp - lse;
}

// ---------------------------------------------------------------------------
// Host launcher
// ---------------------------------------------------------------------------
extern "C" void kernel_launch(void* const* d_in, const int* in_sizes, int n_in,
                              void* d_out, int out_size, void* d_ws, size_t ws_size,
                              hipStream_t stream) {
    const float* rep  = (const float*)d_in[0];  // [B, D]
    const float* mix  = (const float*)d_in[1];  // [C, K]
    const float* loc  = (const float*)d_in[2];  // [C, K, D]
    const float* tril = (const float*)d_in[3];  // [C, K, D, D]
    float* out = (float*)d_out;                 // [B, C]

    // workspace carve-out (256B aligned segments); total ~20.4 MB
    char* ws = (char*)d_ws;
    size_t off = 0;
    auto carve = [&](size_t bytes) -> void* {
        void* p = ws + off;
        off = (off + bytes + 255) & ~(size_t)255;
        return p;
    };
    bf16_t* LinvH  = (bf16_t*)carve((size_t)CK_DIM * D_DIM * D_DIM * sizeof(bf16_t));
    bf16_t* LinvL  = (bf16_t*)carve((size_t)CK_DIM * D_DIM * D_DIM * sizeof(bf16_t));
    bf16_t* Xh     = (bf16_t*)carve((size_t)B_DIM * D_DIM * sizeof(bf16_t));
    bf16_t* Xl     = (bf16_t*)carve((size_t)B_DIM * D_DIM * sizeof(bf16_t));
    float*  wv     = (float*) carve((size_t)CK_DIM * D_DIM * sizeof(float));
    float*  logdet = (float*) carve((size_t)CK_DIM * sizeof(float));
    float*  cl     = (float*) carve((size_t)B_DIM * CK_DIM * sizeof(float));
    (void)ws_size; (void)in_sizes; (void)n_in; (void)out_size;

    const int nX = B_DIM * D_DIM;
    splitx_kernel<<<(nX + 255) / 256, 256, 0, stream>>>(rep, Xh, Xl, nX);
    prep_kernel<<<CK_DIM, 64, 0, stream>>>(tril, loc, LinvH, LinvL, wv, logdet);
    gemm_kernel<<<dim3(CK_DIM, B_DIM / BTILE), 256, 0, stream>>>(
        Xh, Xl, LinvH, LinvL, wv, logdet, cl);
    finalize_kernel<<<B_DIM, 128, 0, stream>>>(mix, cl, out);
}
